// PatchedQwen3_5MoeExperts_59047210385723
// MI455X (gfx1250) — matrix operand order repaired and checked by reading, not compiled, not run on hardware
//
#include <hip/hip_runtime.h>
#include <hip/hip_bf16.h>
#include <math.h>

// ---------------------------------------------------------------------------
// MoE experts (Qwen3.5-style) for MI455X / gfx1250, wave32 + WMMA bf16.
//   E=32, T=2048, H=1024, I=512, K=4
//
// Grid (main): (T/64 token tiles, E/8 expert groups), 256 threads = 8 waves.
// Each block: 64 tokens (4 M-tiles), loops 8 experts with persistent WMMA
// accumulators, merges groups via global fp32 atomics.
// Weights are pre-converted fp32->bf16 in workspace when it is large enough
// (removes all conversion VALU from the hot loop and halves weight bytes).
// ---------------------------------------------------------------------------

#define E_ 32
#define T_ 2048
#define H_ 1024
#define I_ 512
#define K_ 4

typedef __attribute__((ext_vector_type(16))) __bf16 v16bf;
typedef __attribute__((ext_vector_type(8)))  float  v8f;

// --- A-fragment loader (16-bit A 16x32 layout): lane m=lane&15, g=lane>>4.
// lane holds 16 consecutive K = g*16 + {0..15} (matches B).
__device__ __forceinline__ v16bf load_a_lds(const __bf16* s, int stride,
                                            int kbase, int lane) {
  const int m = lane & 15;
  const int g = lane >> 4;
  const __bf16* p = s + m * stride + kbase + g * 16;
  union { v16bf v; uint4 q[2]; } u;
  u.q[0] = *(const uint4*)(p);
  u.q[1] = *(const uint4*)(p + 8);
  return u.v;
}

// --- B-fragment loader (32x16 bf16 B layout): lane n=lane&15 is the output
// column, g=lane>>4 selects the K half; lane holds 16 consecutive K values.
template <bool PRECVT>
__device__ __forceinline__ v16bf load_b(const void* base, size_t row,
                                        int rowStride, int k) {
  if constexpr (PRECVT) {
    const __bf16* p = (const __bf16*)base + row * (size_t)rowStride + k;
    union { v16bf v; uint4 q[2]; } u;
    u.q[0] = *(const uint4*)(p);
    u.q[1] = *(const uint4*)(p + 8);
    return u.v;
  } else {
    const float* p = (const float*)base + row * (size_t)rowStride + k;
    union { v16bf v; __bf16 h[16]; } u;
#pragma unroll
    for (int i = 0; i < 16; i += 4) {
      float4 f = *(const float4*)(p + i);
      u.h[i + 0] = (__bf16)f.x;
      u.h[i + 1] = (__bf16)f.y;
      u.h[i + 2] = (__bf16)f.z;
      u.h[i + 3] = (__bf16)f.w;
    }
    return u.v;
  }
}

// ---------------------------------------------------------------------------
// Prep: fp32 -> bf16 (8 elems / thread: 2x b128 load, 1x b128 store)
// ---------------------------------------------------------------------------
__global__ void cvt_f32_bf16_kernel(const float* __restrict__ x,
                                    __bf16* __restrict__ y, long n8) {
  long i = (long)blockIdx.x * blockDim.x + threadIdx.x;
  if (i >= n8) return;
  const float4 f0 = *(const float4*)(x + i * 8);
  const float4 f1 = *(const float4*)(x + i * 8 + 4);
  union { __bf16 h[8]; uint4 q; } o;
  o.h[0] = (__bf16)f0.x; o.h[1] = (__bf16)f0.y;
  o.h[2] = (__bf16)f0.z; o.h[3] = (__bf16)f0.w;
  o.h[4] = (__bf16)f1.x; o.h[5] = (__bf16)f1.y;
  o.h[6] = (__bf16)f1.z; o.h[7] = (__bf16)f1.w;
  *(uint4*)(y + i * 8) = o.q;
}

// ---------------------------------------------------------------------------
// Prep: densify top-k routing weights -> wtok[T][E]
// ---------------------------------------------------------------------------
__global__ void wtok_kernel(const int* __restrict__ top_k_index,
                            const float* __restrict__ top_k_weights,
                            float* __restrict__ wtok) {
  int t = blockIdx.x * blockDim.x + threadIdx.x;
  if (t >= T_) return;
  float acc[E_];
#pragma unroll
  for (int e = 0; e < E_; ++e) acc[e] = 0.0f;
#pragma unroll
  for (int k = 0; k < K_; ++k) {
    int e = top_k_index[t * K_ + k];
    acc[e] += top_k_weights[t * K_ + k];
  }
#pragma unroll
  for (int e = 0; e < E_; ++e) wtok[t * E_ + e] = acc[e];
}

__global__ void zero_out_kernel(float* __restrict__ out, int n) {
  int i = blockIdx.x * blockDim.x + threadIdx.x;
  if (i < n) out[i] = 0.0f;
}

// ---------------------------------------------------------------------------
// Main MoE kernel.  Block: 256 threads = 8 waves; 64 tokens (4 M-tiles).
//  Stage 1 (per expert): GU = X @ Wgup^T.  Wave w owns inter cols
//    [w*64, w*64+64), split into 2 passes of 32 cols (gate+up pairs) to
//    bound register pressure; each pass: 16 accumulator tiles over K=1024.
//  h = silu(gate)*up*w_tok[row] folded per-row, stored bf16 -> LDS.
//  Stage 2: OUT += h @ Wd^T.  Wave w owns H cols [w*128, w*128+128);
//    persistent c2[8 ntiles][4 mtiles] across the 8-expert loop.
//  Each B fragment is reused by 4 WMMAs (one per M-tile): ~0.75 mem ops/WMMA.
// ---------------------------------------------------------------------------
template <bool PRECVT>
__global__ __launch_bounds__(256)
void moe_wmma_kernel(const __bf16* __restrict__ xb,
                     const float* __restrict__ wtok,
                     const void* __restrict__ gup_w,
                     const void* __restrict__ dwn_w,
                     float* __restrict__ out) {
  __shared__ __attribute__((aligned(16))) __bf16 sX[64 * H_];  // 128 KB
  __shared__ __attribute__((aligned(16))) __bf16 sH[64 * I_];  //  64 KB
  __shared__ float sW[64 * 8];                                 //   2 KB

  const int tokBase = blockIdx.x * 64;
  const int eg      = blockIdx.y;        // experts [eg*8, eg*8+8)
  const int tid     = threadIdx.x;
  const int wave    = tid >> 5;
  const int lane    = tid & 31;
  const int n16     = lane & 15;
  const int g       = lane >> 4;

  const size_t wElem = PRECVT ? sizeof(__bf16) : sizeof(float);

  // --- stage 0: stage X tile (64 x 1024 bf16 = 8192 uint4) into LDS
  {
    const uint4* src = (const uint4*)(xb + (size_t)tokBase * H_);
    uint4* dst = (uint4*)sX;
#pragma unroll
    for (int i = 0; i < 32; ++i) dst[tid + 256 * i] = src[tid + 256 * i];
  }
#pragma unroll
  for (int j = 0; j < 2; ++j) {
    int idx = tid + 256 * j;
    int m = idx >> 3, ee = idx & 7;
    sW[idx] = wtok[(size_t)(tokBase + m) * E_ + eg * 8 + ee];
  }
  __syncthreads();

  const v8f vzero = {};
  v8f c2[8][4];                          // 256 VGPRs persistent
#pragma unroll
  for (int u = 0; u < 8; ++u)
#pragma unroll
    for (int m = 0; m < 4; ++m) c2[u][m] = vzero;

  for (int ee = 0; ee < 8; ++ee) {
    const int e = eg * 8 + ee;
    const char* gup = (const char*)gup_w + (size_t)e * (2 * I_) * H_ * wElem;
    const char* dwn = (const char*)dwn_w + (size_t)e * H_ * I_ * wElem;
    if (ee < 7) {   // hint next expert's slab toward the caches
      __builtin_prefetch((const char*)gup_w +
                         (size_t)(e + 1) * (2 * I_) * H_ * wElem, 0, 1);
    }

    // ---------------- stage 1: gate/up GEMM (K = H = 1024) ----------------
#pragma unroll 1
    for (int p = 0; p < 2; ++p) {        // two 32-col passes
      v8f acc[4][4];                     // [q: 2 gate + 2 up][mtile]
#pragma unroll
      for (int q = 0; q < 4; ++q)
#pragma unroll
        for (int m = 0; m < 4; ++m) acc[q][m] = vzero;

      const int cbase = wave * 64 + p * 32;
      const size_t rowq[4] = {
          (size_t)(cbase + n16),        (size_t)(cbase + 16 + n16),
          (size_t)(I_ + cbase + n16),   (size_t)(I_ + cbase + 16 + n16)};

      for (int ks = 0; ks < 32; ++ks) {
        v16bf a[4];
#pragma unroll
        for (int m = 0; m < 4; ++m)
          a[m] = load_a_lds(sX + m * 16 * H_, H_, ks * 32, lane);
#pragma unroll
        for (int q = 0; q < 4; ++q) {
          v16bf b = load_b<PRECVT>(gup, rowq[q], H_, ks * 32 + g * 16);
#pragma unroll
          for (int m = 0; m < 4; ++m)
            acc[q][m] = __builtin_amdgcn_wmma_f32_16x16x32_bf16(
                false, a[m], false, b, (short)0, acc[q][m], false, false);
        }
      }

      // h = silu(gate) * up * w_tok[row]  -> LDS (bf16)
#pragma unroll
      for (int q = 0; q < 2; ++q) {
#pragma unroll
        for (int m = 0; m < 4; ++m) {
#pragma unroll
          for (int r = 0; r < 8; ++r) {
            int row = m * 16 + r + 8 * g;
            float gv = acc[q][m][r];
            float uv = acc[q + 2][m][r];
            float hv = (gv / (1.0f + __expf(-gv))) * uv * sW[row * 8 + ee];
            sH[row * I_ + cbase + q * 16 + n16] = (__bf16)hv;
          }
        }
      }
    }
    __syncthreads();   // h fully written before any wave reads it

    // ---------------- stage 2: down GEMM (K = I = 512) ----------------
    for (int ks = 0; ks < 16; ++ks) {
      v16bf a[4];
#pragma unroll
      for (int m = 0; m < 4; ++m)
        a[m] = load_a_lds(sH + m * 16 * I_, I_, ks * 32, lane);
#pragma unroll
      for (int u = 0; u < 8; ++u) {
        v16bf b = load_b<PRECVT>(dwn, (size_t)(wave * 128 + u * 16 + n16), I_,
                                 ks * 32 + g * 16);
#pragma unroll
        for (int m = 0; m < 4; ++m)
          c2[u][m] = __builtin_amdgcn_wmma_f32_16x16x32_bf16(
              false, a[m], false, b, (short)0, c2[u][m], false, false);
      }
    }
    __syncthreads();   // all waves done reading sH before next expert
  }

  // ---- merge into output: C layout (row = r + 8*g, col = lane&15) ----
#pragma unroll
  for (int u = 0; u < 8; ++u) {
#pragma unroll
    for (int m = 0; m < 4; ++m) {
#pragma unroll
      for (int r = 0; r < 8; ++r) {
        float* p = out + (size_t)(tokBase + m * 16 + r + 8 * g) * H_
                       + wave * 128 + u * 16 + n16;
        unsafeAtomicAdd(p, c2[u][m][r]);
      }
    }
  }
}

// ---------------------------------------------------------------------------
// Launch
// ---------------------------------------------------------------------------
extern "C" void kernel_launch(void* const* d_in, const int* in_sizes, int n_in,
                              void* d_out, int out_size, void* d_ws, size_t ws_size,
                              hipStream_t stream) {
  const float* hidden_states = (const float*)d_in[0];
  const int*   top_k_index   = (const int*)d_in[1];
  const float* top_k_weights = (const float*)d_in[2];
  const float* gate_up_proj  = (const float*)d_in[3];
  const float* down_proj     = (const float*)d_in[4];
  float*       out           = (float*)d_out;

  const size_t gupBytes = (size_t)E_ * 2 * I_ * H_ * sizeof(__bf16); // 64 MiB
  const size_t dwnBytes = (size_t)E_ * H_ * I_ * sizeof(__bf16);     // 32 MiB
  const size_t xbBytes  = (size_t)T_ * H_ * sizeof(__bf16);          //  4 MiB
  const size_t wtBytes  = (size_t)T_ * E_ * sizeof(float);           // 256 KiB
  const size_t needPre  = gupBytes + dwnBytes + xbBytes + wtBytes;

  dim3 grid(T_ / 64, E_ / 8);

  if (ws_size >= needPre) {
    __bf16* gupB = (__bf16*)d_ws;
    __bf16* dwnB = (__bf16*)((char*)d_ws + gupBytes);
    __bf16* xb   = (__bf16*)((char*)d_ws + gupBytes + dwnBytes);
    float*  wtok = (float*)((char*)d_ws + gupBytes + dwnBytes + xbBytes);

    long n8g = (long)E_ * 2 * I_ * H_ / 8;
    cvt_f32_bf16_kernel<<<(unsigned)((n8g + 255) / 256), 256, 0, stream>>>(
        gate_up_proj, gupB, n8g);
    long n8d = (long)E_ * H_ * I_ / 8;
    cvt_f32_bf16_kernel<<<(unsigned)((n8d + 255) / 256), 256, 0, stream>>>(
        down_proj, dwnB, n8d);
    long n8x = (long)T_ * H_ / 8;
    cvt_f32_bf16_kernel<<<(unsigned)((n8x + 255) / 256), 256, 0, stream>>>(
        hidden_states, xb, n8x);
    wtok_kernel<<<(T_ + 255) / 256, 256, 0, stream>>>(top_k_index,
                                                      top_k_weights, wtok);
    zero_out_kernel<<<(T_ * H_ + 255) / 256, 256, 0, stream>>>(out, T_ * H_);

    moe_wmma_kernel<true><<<grid, 256, 0, stream>>>(xb, wtok, gupB, dwnB, out);
  } else {
    __bf16* xb   = (__bf16*)d_ws;
    float*  wtok = (float*)((char*)d_ws + xbBytes);

    long n8x = (long)T_ * H_ / 8;
    cvt_f32_bf16_kernel<<<(unsigned)((n8x + 255) / 256), 256, 0, stream>>>(
        hidden_states, xb, n8x);
    wtok_kernel<<<(T_ + 255) / 256, 256, 0, stream>>>(top_k_index,
                                                      top_k_weights, wtok);
    zero_out_kernel<<<(T_ * H_ + 255) / 256, 256, 0, stream>>>(out, T_ * H_);

    moe_wmma_kernel<false><<<grid, 256, 0, stream>>>(xb, wtok, gate_up_proj,
                                                     down_proj, out);
  }
}
